// Model_65111704207477
// MI455X (gfx1250) — matrix operand order repaired and checked by reading, not compiled, not run on hardware
//
#include <hip/hip_runtime.h>
#include <stdint.h>

// ---------------------------------------------------------------------------
// MI455X / gfx1250 implementation.
//
//   1) mlp6_kernel        : 6 coefficient MLPs (64->1024->1024->1024->2), f32 VALU
//   2) prep_weights       : f32 (O,I,3,3,3) conv weights -> f16, pre-swizzled into
//                           the exact V_WMMA 16x16x32 A-fragment lane/VGPR layout
//   3) per resnet (u,v):
//        stem_kernel      : 1->32ch conv + ReLU, writes f16 channels-last
//        conv32_wmma x4   : 32->32ch 3^3 conv via implicit GEMM on
//                           v_wmma_f32_16x16x32_f16, with DOUBLE-BUFFERED
//                           global_load_async_to_lds_b128 staging (ASYNCcnt)
//        head_kernel      : 32->1ch conv, f32 out (up / vp slices of d_out)
//   4) sweep_kernel x20   : Jacobi relaxation, ping-pong, final sweep -> u0 slice
//
// d_out layout (f32): [ u0 (N) | up (N) | vp (N) | params_all (12) ], N = 16*32^3.
// ---------------------------------------------------------------------------

typedef __attribute__((ext_vector_type(16))) _Float16 v16h;
typedef __attribute__((ext_vector_type(8)))  float    v8f;
typedef unsigned int u32;

#define NVOX   (16 * 32 * 32 * 32)          // 524288 voxels (B * nt * nx * ny)
#define DWPL   13824                         // wfrag dwords per conv layer: 27*2*32*8

// Recover the 32-bit LDS byte offset from a generic pointer into __shared__.
#define LDS_OFF(p) ((u32)(unsigned long long)                                  \
    (__attribute__((address_space(3))) const void*)(const void*)(p))

// ------------------------------ MLP (6 nets) -------------------------------
__global__ __launch_bounds__(256)
void mlp6_kernel(const float* __restrict__ W0, const float* __restrict__ B0,
                 const float* __restrict__ W1, const float* __restrict__ B1,
                 const float* __restrict__ W2, const float* __restrict__ B2,
                 const float* __restrict__ W3, const float* __restrict__ B3,
                 const float* __restrict__ pin,
                 float* __restrict__ prm_ws, float* __restrict__ prm_out)
{
    __shared__ float hA[1024];
    __shared__ float hB[1024];
    const int k = blockIdx.x;          // which of the 6 nets
    const int tid = threadIdx.x;

    if (tid < 64) hA[tid] = pin[k * 64 + tid];
    __syncthreads();

    for (int o = tid; o < 1024; o += 256) {              // 64 -> 1024, ReLU
        float a = B0[k * 1024 + o];
        const float* w = W0 + (k * 64) * 1024 + o;
        for (int i = 0; i < 64; ++i) a = fmaf(hA[i], w[i * 1024], a);
        hB[o] = fmaxf(a, 0.f);
    }
    __syncthreads();
    for (int o = tid; o < 1024; o += 256) {              // 1024 -> 1024, ReLU
        float a = B1[k * 1024 + o];
        const float* w = W1 + (k * 1024) * 1024 + o;
        for (int i = 0; i < 1024; ++i) a = fmaf(hB[i], w[i * 1024], a);
        hA[o] = fmaxf(a, 0.f);
    }
    __syncthreads();
    for (int o = tid; o < 1024; o += 256) {              // 1024 -> 1024, ReLU
        float a = B2[k * 1024 + o];
        const float* w = W2 + (k * 1024) * 1024 + o;
        for (int i = 0; i < 1024; ++i) a = fmaf(hA[i], w[i * 1024], a);
        hB[o] = fmaxf(a, 0.f);
    }
    __syncthreads();
    if (tid < 2) {                                       // 1024 -> 2, linear
        float a = B3[k * 2 + tid];
        const float* w = W3 + (k * 1024) * 2 + tid;
        for (int i = 0; i < 1024; ++i) a = fmaf(hB[i], w[i * 2], a);
        prm_ws[k * 2 + tid]  = a;
        prm_out[k * 2 + tid] = a;
    }
}

// --------------------- weight pre-swizzle into A-fragments -----------------
// A-fragment layout for V_WMMA_F32_16X16X32_F16 (16-bit A, 16x32, wave32):
//   lane L: g = L>>4, m = (L&15); VGPR v holds K pair:
//     kbase(v) = (v<4) ? 2v : 16 + 2(v-4);  K = kbase + p + 8*g,  p in {0,1}
// Source f32 weights are (Cout=32, Cin=32, 3,3,3): idx = (m*32 + cin)*27 + tap.
struct WPtrs { const float* p[8]; };

__global__ __launch_bounds__(256)
void prep_weights(WPtrs wp, u32* __restrict__ wfrag)
{
    const int f = blockIdx.x * 256 + threadIdx.x;        // 0 .. 8*13824-1 exact
    const int layer = f / DWPL;
    const int rem   = f - layer * DWPL;
    const int v     = rem & 7;
    const int lane  = (rem >> 3) & 31;
    const int mtile = (rem >> 8) & 1;
    const int tap   = rem >> 9;                          // 0..26
    const int g = lane >> 4, mr = lane & 15;
    const int m  = mtile * 16 + mr;
    const int kb = ((v < 4) ? 2 * v : 16 + 2 * (v - 4)) + 8 * g;

    const float* w = wp.p[layer];
    const float w0 = w[(m * 32 + kb)     * 27 + tap];
    const float w1 = w[(m * 32 + kb + 1) * 27 + tap];
    union { _Float16 h[2]; u32 u; } pk;
    pk.h[0] = (_Float16)w0;
    pk.h[1] = (_Float16)w1;
    wfrag[f] = pk.u;
}

// ------------------------------- stem conv ---------------------------------
// 1->32 ch, SAME, + bias + ReLU. Output f16 channels-last [b][z][y][x][32].
__global__ __launch_bounds__(256)
void stem_kernel(const float* __restrict__ u, const float* __restrict__ w,
                 const float* __restrict__ bias, _Float16* __restrict__ out)
{
    __shared__ float ws[32 * 27];
    __shared__ float bs[32];
    const int tid = threadIdx.x;
    for (int i = tid; i < 864; i += 256) ws[i] = w[i];
    if (tid < 32) bs[tid] = bias[tid];
    __syncthreads();

    const int idx = blockIdx.x * 256 + tid;              // voxel id, exact grid
    const int x = idx & 31, y = (idx >> 5) & 31, z = (idx >> 10) & 31, b = idx >> 15;

    float acc[32];
    for (int c = 0; c < 32; ++c) acc[c] = bs[c];
    for (int t = 0; t < 27; ++t) {
        const int dz = t / 9, dy = (t / 3) % 3, dx = t % 3;
        const int zz = z + dz - 1, yy = y + dy - 1, xx = x + dx - 1;
        if ((unsigned)zz < 32u && (unsigned)yy < 32u && (unsigned)xx < 32u) {
            const float val = u[((b * 32 + zz) * 32 + yy) * 32 + xx];
            for (int c = 0; c < 32; ++c) acc[c] = fmaf(ws[c * 27 + t], val, acc[c]);
        }
    }
    _Float16* o = out + idx * 32;                        // 64B contiguous per voxel
    union { _Float16 h[8]; uint4 q; } pk;
    for (int c8 = 0; c8 < 4; ++c8) {
        for (int j = 0; j < 8; ++j) pk.h[j] = (_Float16)fmaxf(acc[c8 * 8 + j], 0.f);
        *(uint4*)(o + c8 * 8) = pk.q;
    }
}

// ------------------------- 32->32 conv via WMMA ----------------------------
// Implicit GEMM: M=32 out-ch, K=27x32, N=32 voxels (one x-row per workgroup).
// 4 waves: wave w -> (Mtile=w&1, Ntile=w>>1). Per (dz,dy) a padded 34-voxel
// channels-last row is staged in LDS via global_load_async_to_lds_b128 into a
// DOUBLE-BUFFERED pair of rows: while row dzy is consumed by 3 WMMAs, row
// dzy+1 is already in flight on the async path (ASYNCcnt). OOB rows issue a
// clamped-address copy so every iteration has exactly one outstanding async
// op per wave (uniform s_wait_asynccnt immediates); their compute is skipped
// by a block-uniform branch, so EXEC is all-ones around every WMMA.
// B-fragment assumption (K=32 x N=16, 16-bit, wave32): lane (g,n) holds
// K = 8*g + h (h<8) or 16 + 8*g + (h-8) (h>=8), n = lane&15 -- same k order
// as the prep_weights A-fragments (AMD table order).
__device__ __forceinline__
void async_row_copy(const _Float16* __restrict__ in, _Float16* lds_dst,
                    int b, int zzc, int yyc, int vx, int c8)
{
    const _Float16* src = in + (((b * 32 + zzc) * 32 + yyc) * 32 + vx) * 32 + c8;
    const u32 ldsoff = LDS_OFF(lds_dst);
    asm volatile("global_load_async_to_lds_b128 %0, %1, off"
                 :: "v"(ldsoff), "v"((unsigned long long)src)
                 : "memory");
}

__global__ __launch_bounds__(128)
void conv32_wmma(const _Float16* __restrict__ in, const _Float16* __restrict__ res,
                 _Float16* __restrict__ out, const u32* __restrict__ wfrag,
                 const float* __restrict__ bias, int add_res)
{
    __shared__ _Float16 rowbuf0[34 * 32];                // 2176 B each
    __shared__ _Float16 rowbuf1[34 * 32];
    const int wg = blockIdx.x;                           // 16*32*32 rows
    const int y = wg & 31, z = (wg >> 5) & 31, b = wg >> 10;
    const int tid  = threadIdx.x;
    const int lane = tid & 31, wave = tid >> 5;
    const int mtile = wave & 1, ntile = wave >> 1;
    const int g = lane >> 4, ln = lane & 15;
    const int vx = tid >> 2, c8 = (tid & 3) * 8;         // 128 thr x 16B = 2KB/row

    // Zero the x = -1 / x = 32 pad voxels of both buffers once (never restaged).
    if (tid < 16) {
        ((u32*)rowbuf0)[tid] = 0u;
        ((u32*)rowbuf1)[tid] = 0u;
    } else if (tid < 32) {
        ((u32*)rowbuf0)[33 * 16 + (tid - 16)] = 0u;
        ((u32*)rowbuf1)[33 * 16 + (tid - 16)] = 0u;
    }

    // Kick off row dzy = 0 (dz = dy = 0 -> zz = z-1, yy = y-1, clamped).
    {
        const int zz = z - 1, yy = y - 1;
        const int zzc = zz < 0 ? 0 : (zz > 31 ? 31 : zz);
        const int yyc = yy < 0 ? 0 : (yy > 31 ? 31 : yy);
        async_row_copy(in, rowbuf0 + (vx + 1) * 32 + c8, b, zzc, yyc, vx, c8);
    }

    v8f acc = {};
    for (int dzy = 0; dzy < 9; ++dzy) {
        _Float16* cur = (dzy & 1) ? rowbuf1 : rowbuf0;
        _Float16* nxt = (dzy & 1) ? rowbuf0 : rowbuf1;
        if (dzy < 8) {                                   // prefetch row dzy+1
            const int dz1 = (dzy + 1) / 3, dy1 = (dzy + 1) - dz1 * 3;
            const int zz = z + dz1 - 1, yy = y + dy1 - 1;
            const int zzc = zz < 0 ? 0 : (zz > 31 ? 31 : zz);
            const int yyc = yy < 0 ? 0 : (yy > 31 ? 31 : yy);
            async_row_copy(in, nxt + (vx + 1) * 32 + c8, b, zzc, yyc, vx, c8);
            asm volatile("s_wait_asynccnt 0x1" ::: "memory"); // row dzy landed
        } else {
            asm volatile("s_wait_asynccnt 0x0" ::: "memory");
        }
        __syncthreads();                                 // all waves' row dzy visible

        const int dz = dzy / 3, dy = dzy - dz * 3;
        const int zz = z + dz - 1, yy = y + dy - 1;
        const bool rowok = ((unsigned)zz < 32u) && ((unsigned)yy < 32u);
        if (rowok) {                                     // block-uniform branch
            for (int dx = 0; dx < 3; ++dx) {
                const int tap = dzy * 3 + dx;
                const u32* wpw = wfrag + ((tap * 2 + mtile) * 32 + lane) * 8;
                v16h a = *(const v16h*)wpw;              // 32B A-fragment (L2-hot)
                const _Float16* bp = cur + (ntile * 16 + ln + dx) * 32 + g * 8;
                typedef __attribute__((ext_vector_type(8))) _Float16 v8h_t;
                const v8h_t blo = *(const v8h_t*)bp;
                const v8h_t bhi = *(const v8h_t*)(bp + 16);
                v16h bf = __builtin_shufflevector(blo, bhi, 0,1,2,3,4,5,6,7,8,9,10,11,12,13,14,15); // 2x ds_load_b128
                acc = __builtin_amdgcn_wmma_f32_16x16x32_f16(
                          false, a, false, bf, (short)0, acc, false, false);
            }
        }
        __syncthreads();                                 // readers done before cur
    }                                                    // is overwritten next iter

    // Epilogue: C/D layout -> lane (g,ln), VGPR r: m = mtile*16 + 8*g + r (consecutive!)
    const int m0  = mtile * 16 + g * 8;
    const int x   = ntile * 16 + ln;
    const int vox = (((b * 32 + z) * 32 + y) * 32 + x) * 32;
    float bv[8];
    *(float4*)&bv[0] = *(const float4*)(bias + m0);
    *(float4*)&bv[4] = *(const float4*)(bias + m0 + 4);
    union { _Float16 h[8]; uint4 q; } rs, ot;
    if (add_res) rs.q = *(const uint4*)(res + vox + m0);
    for (int r = 0; r < 8; ++r) {
        float f = acc[r] + bv[r];
        if (add_res) f += (float)rs.h[r];
        ot.h[r] = (_Float16)fmaxf(f, 0.f);
    }
    *(uint4*)(out + vox + m0) = ot.q;                    // one b128 store
}

// ------------------------------- head conv ---------------------------------
// 32->1 ch + bias (no ReLU). f16 channels-last in, f32 out.
__global__ __launch_bounds__(256)
void head_kernel(const _Float16* __restrict__ h, const float* __restrict__ w,
                 const float* __restrict__ bias, float* __restrict__ out)
{
    __shared__ float ws[32 * 27];
    __shared__ float b0;
    const int tid = threadIdx.x;
    for (int i = tid; i < 864; i += 256) ws[i] = w[i];
    if (tid == 0) b0 = bias[0];
    __syncthreads();

    const int idx = blockIdx.x * 256 + tid;
    const int x = idx & 31, y = (idx >> 5) & 31, z = (idx >> 10) & 31, b = idx >> 15;
    float acc = b0;
    for (int t = 0; t < 27; ++t) {
        const int dz = t / 9, dy = (t / 3) % 3, dx = t % 3;
        const int zz = z + dz - 1, yy = y + dy - 1, xx = x + dx - 1;
        if ((unsigned)zz < 32u && (unsigned)yy < 32u && (unsigned)xx < 32u) {
            const _Float16* hp = h + (((b * 32 + zz) * 32 + yy) * 32 + xx) * 32;
            for (int c = 0; c < 32; ++c) acc = fmaf((float)hp[c], ws[c * 27 + t], acc);
        }
    }
    out[idx] = acc;
}

// ---------------------------- Jacobi relaxation ----------------------------
// One sweep: interior updated from src, boundary passed through (Dirichlet=up).
// Coefficient fields recomputed from up/vp each sweep (stays in L2).
__global__ __launch_bounds__(256)
void sweep_kernel(const float* __restrict__ src, float* __restrict__ dst,
                  const float* __restrict__ upf, const float* __restrict__ vpf,
                  const float* __restrict__ prm)
{
    const int idx = blockIdx.x * 256 + threadIdx.x;      // exact: NVOX threads
    const int iy = idx & 31, ix = (idx >> 5) & 31, it = (idx >> 10) & 31;
    const float cur = src[idx];
    const bool interior = it > 0 && it < 31 && ix > 0 && ix < 31 && iy > 0 && iy < 31;
    if (!interior) { dst[idx] = cur; return; }

    const float p00 = prm[0];   // params_all[0,0]
    const float p20 = prm[4];   // params_all[2,0] = c6
    const float p30 = prm[6];   // params_all[3,0] = c5
    const float p40 = prm[8];   // params_all[4,0]
    const float p50 = prm[10];  // params_all[5,0]

    const float DT = 0.05f, DX = 0.35f, DY = 0.35f, OMEGA = 0.6f;
    const float upv = upf[idx], vpv = vpf[idx];
    const float A2  = upv * upv + vpv * vpv;
    const float c0  = p50 + p00 * A2;
    const float rhs = p40 * A2 * vpv;
    const float c5 = p30, c6 = p20;
    float diag = c0 - 2.f * c5 / (DX * DX) - 2.f * c6 / (DY * DY);
    diag += (diag >= 0.f) ? 1e-3f : -1e-3f;

    const float ut  = (src[idx + 1024] - src[idx - 1024]) * (0.5f / DT);
    const float uxx = (src[idx + 32] - 2.f * cur + src[idx - 32]) / (DX * DX);
    const float uyy = (src[idx + 1]  - 2.f * cur + src[idx - 1])  / (DY * DY);
    const float r   = rhs - (c0 * cur + ut + c5 * uxx + c6 * uyy);
    dst[idx] = cur + OMEGA * r / diag;
}

// ------------------------------- launcher ----------------------------------
extern "C" void kernel_launch(void* const* d_in, const int* in_sizes, int n_in,
                              void* d_out, int out_size, void* d_ws, size_t ws_size,
                              hipStream_t stream)
{
    // d_in order (tree-flattened setup_inputs):
    // 0:u 1:v 2:t 3:x 4:y
    // 5..16  : rnet1 (ws,bs, w1,b1,w2,b2, w3,b3,w4,b4, wh,bh)
    // 17..28 : rnet2 (same)
    // 29..36 : mlp (W0,b0,W1,b1,W2,b2,W3,b3)   37: param_ins
    const float* u   = (const float*)d_in[0];
    const float* v   = (const float*)d_in[1];
    const float* pin = (const float*)d_in[37];

    // Workspace layout (~70 MB)
    char* ws = (char*)d_ws;
    _Float16* hbuf = (_Float16*)(ws);                    // 32 MiB f16 activations
    _Float16* rbuf = (_Float16*)(ws + 33554432);         // 32 MiB f16 activations
    u32*   wfrag   = (u32*)  (ws + 67108864);            // 8 * 13824 dwords
    float* prm     = (float*)(ws + 67551232);            // 12 floats
    float* bufB    = (float*)(ws + 67551488);            // 2 MiB sweep ping buffer

    float* out     = (float*)d_out;
    float* u0      = out;
    float* up      = out + NVOX;
    float* vp      = out + 2 * NVOX;
    float* prm_out = out + 3 * NVOX;

    // 1) coefficient MLPs
    mlp6_kernel<<<6, 256, 0, stream>>>(
        (const float*)d_in[29], (const float*)d_in[30],
        (const float*)d_in[31], (const float*)d_in[32],
        (const float*)d_in[33], (const float*)d_in[34],
        (const float*)d_in[35], (const float*)d_in[36],
        pin, prm, prm_out);

    // 2) pre-swizzle the eight 32->32 conv weights into WMMA A-fragments
    WPtrs wp;
    wp.p[0] = (const float*)d_in[7];  wp.p[1] = (const float*)d_in[9];
    wp.p[2] = (const float*)d_in[11]; wp.p[3] = (const float*)d_in[13];
    wp.p[4] = (const float*)d_in[19]; wp.p[5] = (const float*)d_in[21];
    wp.p[6] = (const float*)d_in[23]; wp.p[7] = (const float*)d_in[25];
    prep_weights<<<432, 256, 0, stream>>>(wp, wfrag);

    // 3) the two ResNet3Ds
    for (int s = 0; s < 2; ++s) {
        const int base = 5 + 12 * s;
        const float* x_in = s ? v : u;
        stem_kernel<<<NVOX / 256, 256, 0, stream>>>(
            x_in, (const float*)d_in[base], (const float*)d_in[base + 1], hbuf);
        // block1: r = relu(conv(h)); h = relu(h + conv(r))
        conv32_wmma<<<16384, 128, 0, stream>>>(hbuf, nullptr, rbuf,
            wfrag + (4 * s + 0) * DWPL, (const float*)d_in[base + 3], 0);
        conv32_wmma<<<16384, 128, 0, stream>>>(rbuf, hbuf, hbuf,
            wfrag + (4 * s + 1) * DWPL, (const float*)d_in[base + 5], 1);
        // block2
        conv32_wmma<<<16384, 128, 0, stream>>>(hbuf, nullptr, rbuf,
            wfrag + (4 * s + 2) * DWPL, (const float*)d_in[base + 7], 0);
        conv32_wmma<<<16384, 128, 0, stream>>>(rbuf, hbuf, hbuf,
            wfrag + (4 * s + 3) * DWPL, (const float*)d_in[base + 9], 1);
        head_kernel<<<NVOX / 256, 256, 0, stream>>>(
            hbuf, (const float*)d_in[base + 10], (const float*)d_in[base + 11],
            s ? vp : up);
    }

    // 4) 20 Jacobi sweeps, ping-pong, ending in the u0 slice of d_out
    for (int k = 0; k < 20; ++k) {
        const float* src = (k == 0) ? up : ((k & 1) ? bufB : u0);
        float*       dst = (k & 1) ? u0 : bufB;
        sweep_kernel<<<NVOX / 256, 256, 0, stream>>>(src, dst, up, vp, prm);
    }
}
